// get_model_16681652978027
// MI455X (gfx1250) — matrix operand order repaired and checked by reading, not compiled, not run on hardware
//
#include <hip/hip_runtime.h>
#include <hip/hip_bf16.h>

typedef __attribute__((ext_vector_type(16))) _Float16 v16h;
typedef __attribute__((ext_vector_type(8)))  _Float16 v8h;
typedef __attribute__((ext_vector_type(8)))  float    v8f;

#define BB   8
#define NNP  2048
#define KNN  30
#define NPTS (BB * NNP)

// ---------------------------------------------------------------- x0 = [pos, x]
__global__ void k_build_x0(const float* __restrict__ pos, const float* __restrict__ x,
                           float* __restrict__ x0) {
  int p = blockIdx.x * blockDim.x + threadIdx.x;
  if (p >= NPTS) return;
  for (int c = 0; c < 3; ++c) {
    x0[p * 6 + c]     = pos[p * 3 + c];
    x0[p * 6 + 3 + c] = x[p * 3 + c];
  }
}

// ---------------------------------------------------------------- transpose + cvt weights
// W (K x N, row-major f32) -> Wt (Npad x Kpad, f16), Wt[n][k] = W[k][n], zero padded
__global__ void k_prep_wt(const float* __restrict__ W, int K, int N, int Kpad, int Npad,
                          _Float16* __restrict__ Wt) {
  int t = blockIdx.x * blockDim.x + threadIdx.x;
  if (t >= Kpad * Npad) return;
  int n = t / Kpad, k = t % Kpad;
  float v = (k < K && n < N) ? W[(size_t)k * N + n] : 0.f;
  Wt[t] = (_Float16)v;
}

// ---------------------------------------------------------------- squared norms
__global__ void k_sqnorm(const float* __restrict__ f, int F, float* __restrict__ sq) {
  int p = blockIdx.x * blockDim.x + threadIdx.x;
  if (p >= NPTS) return;
  float s = 0.f;
  for (int c = 0; c < F; ++c) { float v = f[(size_t)p * F + c]; s += v * v; }
  sq[p] = s;
}

// ---------------------------------------------------------------- kNN (top-30 smallest d)
__global__ void k_knn(const float* __restrict__ feat, const float* __restrict__ sq,
                      int F, int* __restrict__ idx) {
  int p = blockIdx.x * blockDim.x + threadIdx.x;
  if (p >= NPTS) return;
  int b = p / NNP;
  float xi[64];
  for (int c = 0; c < F; ++c) xi[c] = feat[(size_t)p * F + c];
  float sqi = sq[p];
  float dist[KNN]; int ind[KNN];
  for (int t = 0; t < KNN; ++t) { dist[t] = 3.4e38f; ind[t] = 0x7fffffff; }
  float wd = 3.4e38f; int wi = 0x7fffffff; int wslot = 0;
  const float* fb  = feat + (size_t)b * NNP * F;
  const float* sqb = sq + (size_t)b * NNP;
  for (int j = 0; j < NNP; ++j) {
    const float* xj = fb + (size_t)j * F;
    float dot = 0.f;
    for (int c = 0; c < F; ++c) dot += xi[c] * xj[c];
    float d = sqi + sqb[j] - 2.0f * dot;
    if (d < wd || (d == wd && j < wi)) {
      dist[wslot] = d; ind[wslot] = j;
      wd = -3.4e38f; wi = -1;
      for (int t = 0; t < KNN; ++t) {
        if (dist[t] > wd || (dist[t] == wd && ind[t] > wi)) {
          wd = dist[t]; wi = ind[t]; wslot = t;
        }
      }
    }
  }
  for (int t = 0; t < KNN; ++t) idx[(size_t)p * KNN + t] = ind[t];
}

// ---- fragment helpers -------------------------------------------------------
__device__ __forceinline__ v16h frag16(const _Float16* base) {
  v8h lo = *(const v8h*)base;
  v8h hi = *(const v8h*)(base + 8);
  v16h a;
  for (int t = 0; t < 8; ++t) { a[t] = lo[t]; a[t + 8] = hi[t]; }
  return a;
}
__device__ __forceinline__ v16h fragA(const _Float16* rowBase, int kb, int hi) {
  // A 16x32 f16 fragment in this code's k order (B matches): lane owns one contiguous 16-half segment (k = 16*hi + s)
  const _Float16* b = rowBase + kb * 32 + hi * 16;
  v8h lo = *(const v8h*)b;
  v8h h8 = *(const v8h*)(b + 8);
  v16h a;
  for (int t = 0; t < 8; ++t) { a[t] = lo[t]; a[t + 8] = h8[t]; }
  return a;
}

// ---------------------------------------------------------------- EdgeConv (one wave = one point)
__global__ __launch_bounds__(128)
void k_edgeconv(const float* __restrict__ feat, int F, const int* __restrict__ idx,
                const _Float16* __restrict__ W1t, int Kpad,   // W1t: 64 x Kpad
                const float* __restrict__ b1,
                const float* __restrict__ g,  const float* __restrict__ be,
                const _Float16* __restrict__ W2t,             // W2t: 64 x 64
                const float* __restrict__ b2,
                float* __restrict__ out) {
  const int wv   = threadIdx.x >> 5;
  const int lane = threadIdx.x & 31;
  const int p    = blockIdx.x * 4 + wv;          // 4096 blocks -> 16384 points exactly
  const int b    = p / NNP;
  const int Kact = 2 * F;

  __shared__ __align__(16) _Float16 e_lds[4][32][128];
  __shared__ __align__(16) _Float16 h1_lds[4][32][64];

  { // build edge feature rows: row kk = [xi, xj - xi]
    int kk = lane;
    _Float16* row = e_lds[wv][kk];
    if (kk < KNN) {
      int j = idx[(size_t)p * KNN + kk];
      const float* xi = feat + (size_t)p * F;
      const float* xj = feat + ((size_t)b * NNP + j) * F;
      for (int c = 0; c < F; ++c) {
        float a = xi[c];
        row[c]     = (_Float16)a;
        row[F + c] = (_Float16)(xj[c] - a);
      }
      for (int c = Kact; c < Kpad; ++c) row[c] = (_Float16)0.f;
    } else {
      for (int c = 0; c < Kpad; ++c) row[c] = (_Float16)0.f;
    }
  }
  __syncthreads();

  const int nl = lane & 15;
  const int hi = (lane >= 16) ? 1 : 0;

  // ---- GEMM1: h1 = relu((e@W1 + b1)*g + be)
  for (int mt = 0; mt < 2; ++mt) {
    int arow = mt * 16 + nl;
    for (int nb = 0; nb < 4; ++nb) {
      int n = nb * 16 + nl;
      const _Float16* wcol = W1t + (size_t)n * Kpad;
      v8f c = {};
      for (int kb = 0; kb * 32 < Kpad; ++kb) {
        v16h a  = fragA(&e_lds[wv][arow][0], kb, hi);
        v16h bf = frag16(wcol + kb * 32 + hi * 16);
        c = __builtin_amdgcn_wmma_f32_16x16x32_f16(false, a, false, bf, (short)0, c, false, false);
      }
      float gg = g[n], bb = b1[n], ee = be[n];
      for (int r = 0; r < 8; ++r) {
        float h = (c[r] + bb) * gg + ee;
        h = h > 0.f ? h : 0.f;
        h1_lds[wv][mt * 16 + r + hi * 8][n] = (_Float16)h;
      }
    }
  }
  __syncthreads();

  // ---- GEMM2: h2 = h1@W2 + b2 ; out = max over valid rows (0..29)
  for (int nb = 0; nb < 4; ++nb) {
    int n = nb * 16 + nl;
    const _Float16* wcol = W2t + (size_t)n * 64;
    float bb = b2[n];
    float m = -3.4e38f;
    for (int mt = 0; mt < 2; ++mt) {
      int arow = mt * 16 + nl;
      v8f c = {};
      for (int kb = 0; kb < 2; ++kb) {
        v16h a  = fragA(&h1_lds[wv][arow][0], kb, hi);
        v16h bf = frag16(wcol + kb * 32 + hi * 16);
        c = __builtin_amdgcn_wmma_f32_16x16x32_f16(false, a, false, bf, (short)0, c, false, false);
      }
      int rmax = (mt == 1 && hi) ? 6 : 8;   // exclude padded edge rows 30,31
      for (int r = 0; r < rmax; ++r) {
        float v = c[r] + bb;
        if (v > m) m = v;
      }
    }
    float o = __shfl_xor(m, 16, 32);
    m = m > o ? m : o;
    if (lane < 16) out[(size_t)p * 64 + n] = m;
  }
}

// ---------------------------------------------------------------- h0 = f16 concat(x1,x2,x3)
__global__ void k_concat_h0(const float* __restrict__ x1, const float* __restrict__ x2,
                            const float* __restrict__ x3, _Float16* __restrict__ h0) {
  int t = blockIdx.x * blockDim.x + threadIdx.x;
  if (t >= NPTS * 192) return;
  int p = t / 192, c = t % 192;
  float v = (c < 64) ? x1[(size_t)p * 64 + c]
          : (c < 128) ? x2[(size_t)p * 64 + (c - 64)]
                      : x3[(size_t)p * 64 + (c - 128)];
  h0[t] = (_Float16)v;
}

// ---------------------------------------------------------------- generic WMMA GEMM (wave = 16x64 tile)
// A: M x K f16 (K % 32 == 0); Wt: Npad x K f16 col-major(original) ; out N = Ncols
__global__ __launch_bounds__(128)
void k_gemm(const _Float16* __restrict__ A, const _Float16* __restrict__ Wt,
            const float* __restrict__ bias, int M, int K, int Ncols, int relu,
            _Float16* __restrict__ outH, float* __restrict__ outF) {
  int wv = threadIdx.x >> 5, lane = threadIdx.x & 31;
  int tiles_n = (Ncols + 63) / 64;
  int tiles_m = M / 16;
  int tile = blockIdx.x * 4 + wv;
  if (tile >= tiles_m * tiles_n) return;
  int tm = tile / tiles_n, tn = tile % tiles_n;
  int nl = lane & 15, hi = (lane >= 16) ? 1 : 0;
  const _Float16* Arow = A + (size_t)(tm * 16 + nl) * K;
  for (int nb = 0; nb < 4; ++nb) {
    int n = tn * 64 + nb * 16 + nl;
    const _Float16* wcol = Wt + (size_t)n * K;
    v8f c = {};
    for (int kb = 0; kb * 32 < K; ++kb) {
      v16h a  = fragA(Arow, kb, hi);
      v16h bf = frag16(wcol + kb * 32 + hi * 16);
      c = __builtin_amdgcn_wmma_f32_16x16x32_f16(false, a, false, bf, (short)0, c, false, false);
    }
    if (n < Ncols) {
      float bb = bias[n];
      for (int r = 0; r < 8; ++r) {
        float v = c[r] + bb;
        if (relu) v = v > 0.f ? v : 0.f;
        size_t o = (size_t)(tm * 16 + r + hi * 8) * Ncols + n;
        if (outH) outH[o] = (_Float16)v;
        if (outF) outF[o] = v;
      }
    }
  }
}

// ---------------------------------------------------------------- log_softmax over 13
__global__ void k_logsoftmax(const float* __restrict__ logits, float* __restrict__ out) {
  int p = blockIdx.x * blockDim.x + threadIdx.x;
  if (p >= NPTS) return;
  float mx = -3.4e38f;
  for (int c = 0; c < 13; ++c) { float v = logits[p * 13 + c]; if (v > mx) mx = v; }
  float s = 0.f;
  for (int c = 0; c < 13; ++c) s += expf(logits[p * 13 + c] - mx);
  float ls = logf(s);
  for (int c = 0; c < 13; ++c) out[p * 13 + c] = logits[p * 13 + c] - mx - ls;
}

// ================================================================ host
extern "C" void kernel_launch(void* const* d_in, const int* in_sizes, int n_in,
                              void* d_out, int out_size, void* d_ws, size_t ws_size,
                              hipStream_t stream) {
  (void)in_sizes; (void)n_in; (void)out_size; (void)ws_size;
  const float* x   = (const float*)d_in[0];
  const float* pos = (const float*)d_in[1];
  // d_in[2] = batch (unused; layout is dense B x N)
  const float* cW1[3] = {(const float*)d_in[3],  (const float*)d_in[9],  (const float*)d_in[15]};
  const float* cb1[3] = {(const float*)d_in[4],  (const float*)d_in[10], (const float*)d_in[16]};
  const float* cg [3] = {(const float*)d_in[5],  (const float*)d_in[11], (const float*)d_in[17]};
  const float* cbe[3] = {(const float*)d_in[6],  (const float*)d_in[12], (const float*)d_in[18]};
  const float* cW2[3] = {(const float*)d_in[7],  (const float*)d_in[13], (const float*)d_in[19]};
  const float* cb2[3] = {(const float*)d_in[8],  (const float*)d_in[14], (const float*)d_in[20]};
  const float* mW[4] = {(const float*)d_in[21], (const float*)d_in[23],
                        (const float*)d_in[25], (const float*)d_in[27]};
  const float* mb[4] = {(const float*)d_in[22], (const float*)d_in[24],
                        (const float*)d_in[26], (const float*)d_in[28]};

  // workspace carve (256B aligned)
  char* base = (char*)d_ws;
  size_t off = 0;
  auto carve = [&](size_t bytes) -> char* {
    char* p = base + off;
    off = (off + bytes + 255) & ~(size_t)255;
    return p;
  };
  float*    x0     = (float*)   carve((size_t)NPTS * 6  * 4);
  float*    sq     = (float*)   carve((size_t)NPTS * 4);
  int*      idx    = (int*)     carve((size_t)NPTS * KNN * 4);
  float*    x1     = (float*)   carve((size_t)NPTS * 64 * 4);
  float*    x2     = (float*)   carve((size_t)NPTS * 64 * 4);
  float*    x3     = (float*)   carve((size_t)NPTS * 64 * 4);
  _Float16* h0     = (_Float16*)carve((size_t)NPTS * 192 * 2);
  _Float16* h1     = (_Float16*)carve((size_t)NPTS * 1024 * 2);
  _Float16* h2     = (_Float16*)carve((size_t)NPTS * 256 * 2);
  _Float16* h3     = (_Float16*)carve((size_t)NPTS * 128 * 2);
  float*    logits = (float*)   carve((size_t)NPTS * 13 * 4);

  // transposed / padded f16 weights
  const int cKpad[3] = {32, 128, 128};           // EdgeConv W1 K-pads (2F rounded to 32)
  _Float16* cW1t[3]; _Float16* cW2t[3];
  for (int L = 0; L < 3; ++L) {
    cW1t[L] = (_Float16*)carve((size_t)64 * cKpad[L] * 2);
    cW2t[L] = (_Float16*)carve((size_t)64 * 64 * 2);
  }
  // MLP: (K, N, Npad)
  const int mK[4] = {192, 1024, 256, 128};
  const int mN[4] = {1024, 256, 128, 13};
  const int mNp[4] = {1024, 256, 128, 64};
  _Float16* mWt[4];
  for (int j = 0; j < 4; ++j) mWt[j] = (_Float16*)carve((size_t)mNp[j] * mK[j] * 2);

  const int TPB = 256;
  const int PB  = (NPTS + TPB - 1) / TPB;

  // prep weights
  for (int L = 0; L < 3; ++L) {
    int e1 = 64 * cKpad[L];
    k_prep_wt<<<(e1 + TPB - 1) / TPB, TPB, 0, stream>>>(cW1[L], 2 * (L ? 64 : 6), 64,
                                                        cKpad[L], 64, cW1t[L]);
    k_prep_wt<<<(64 * 64 + TPB - 1) / TPB, TPB, 0, stream>>>(cW2[L], 64, 64, 64, 64, cW2t[L]);
  }
  for (int j = 0; j < 4; ++j) {
    int e = mNp[j] * mK[j];
    k_prep_wt<<<(e + TPB - 1) / TPB, TPB, 0, stream>>>(mW[j], mK[j], mN[j], mK[j], mNp[j], mWt[j]);
  }

  k_build_x0<<<PB, TPB, 0, stream>>>(pos, x, x0);

  const float* feats[3] = {x0, x1, x2};
  const int    Fs[3]    = {6, 64, 64};
  float*       outs[3]  = {x1, x2, x3};
  for (int L = 0; L < 3; ++L) {
    k_sqnorm<<<PB, TPB, 0, stream>>>(feats[L], Fs[L], sq);
    k_knn   <<<PB, TPB, 0, stream>>>(feats[L], sq, Fs[L], idx);
    k_edgeconv<<<NPTS / 4, 128, 0, stream>>>(feats[L], Fs[L], idx,
                                             cW1t[L], cKpad[L], cb1[L], cg[L], cbe[L],
                                             cW2t[L], cb2[L], outs[L]);
  }

  k_concat_h0<<<(NPTS * 192 + TPB - 1) / TPB, TPB, 0, stream>>>(x1, x2, x3, h0);

  auto gemm_blocks = [](int M, int N) {
    int tiles = (M / 16) * ((N + 63) / 64);
    return (tiles + 3) / 4;
  };
  k_gemm<<<gemm_blocks(NPTS, 1024), 128, 0, stream>>>(h0, mWt[0], mb[0], NPTS, 192, 1024, 1, h1, nullptr);
  k_gemm<<<gemm_blocks(NPTS, 256),  128, 0, stream>>>(h1, mWt[1], mb[1], NPTS, 1024, 256, 1, h2, nullptr);
  k_gemm<<<gemm_blocks(NPTS, 128),  128, 0, stream>>>(h2, mWt[2], mb[2], NPTS, 256, 128, 1, h3, nullptr);
  k_gemm<<<gemm_blocks(NPTS, 13),   128, 0, stream>>>(h3, mWt[3], mb[3], NPTS, 128, 13, 0, nullptr, logits);

  k_logsoftmax<<<PB, TPB, 0, stream>>>(logits, (float*)d_out);

  // second output: x3 (B,N,64) f32, appended after the (B,N,13) log_softmax
  hipMemcpyAsync((float*)d_out + (size_t)NPTS * 13, x3,
                 (size_t)NPTS * 64 * 4, hipMemcpyDeviceToDevice, stream);
}
